// Qwen3MoEBlock_40492951667104
// MI455X (gfx1250) — matrix operand order repaired and checked by reading, not compiled, not run on hardware
//
#include <hip/hip_runtime.h>

// ---------------------------------------------------------------------------
// Qwen3-MoE block for MI455X (gfx1250, wave32, WMMA + TDM).
// B=1, S=512, DIM=2048, H=32, HK=4, HD=128, E=64, TOPK=8, FF=768
// ---------------------------------------------------------------------------

typedef __bf16 bf16;
typedef __attribute__((ext_vector_type(16))) __bf16 v16bf;
typedef __attribute__((ext_vector_type(8)))  __bf16 v8bf;
typedef __attribute__((ext_vector_type(8)))  float  v8f;
typedef __attribute__((ext_vector_type(4)))  float  v4f;
typedef __attribute__((ext_vector_type(4)))  unsigned int u32x4;
typedef __attribute__((ext_vector_type(8)))  int i32x8;
typedef __attribute__((ext_vector_type(4)))  int i32x4;

#define S_LEN 512
#define DIM   2048
#define NH    32
#define NKV   4
#define HD    128
#define NE    64
#define NTOP  8
#define FF    768
#define AK_CHUNK 256   // K-chunk staged to LDS per TDM descriptor

#if defined(__has_builtin)
# if __has_builtin(__builtin_amdgcn_tensor_load_to_lds) && \
     __has_builtin(__builtin_amdgcn_s_wait_tensorcnt)
#  define HAVE_TDM 1
# endif
# if __has_builtin(__builtin_amdgcn_tensor_store_from_lds)
#  define HAVE_TDM_STORE 1
# endif
#endif
#ifndef HAVE_TDM
# define HAVE_TDM 0
#endif
#ifndef HAVE_TDM_STORE
# define HAVE_TDM_STORE 0
#endif

static __device__ __forceinline__ v8f zero8() {
  v8f z;
#pragma unroll
  for (int i = 0; i < 8; ++i) z[i] = 0.0f;
  return z;
}

// A-fragment (16x32 bf16): lane half h holds K in [16h, 16h+16)
static __device__ __forceinline__ v16bf load_afrag(const bf16* rowk, int khalf) {
  v8bf c0 = *(const v8bf*)(rowk + khalf * 16);
  v8bf c1 = *(const v8bf*)(rowk + khalf * 16 + 8);
  v16bf r;
#pragma unroll
  for (int i = 0; i < 8; ++i) { r[i] = c0[i]; r[i + 8] = c1[i]; }
  return r;
}

// B-fragment (32x16 bf16): lane half h holds K in [16h, 16h+16), N = lane&15
static __device__ __forceinline__ v16bf load_bfrag(const bf16* p) {
  return *(const v16bf*)p;
}

// B-fragment from f32 weights: 16 contiguous floats -> bf16 in registers
static __device__ __forceinline__ v16bf load_wfrag(const float* p) {
  v16bf r;
#pragma unroll
  for (int i = 0; i < 16; i += 4) {
    v4f f = *(const v4f*)(p + i);
    r[i + 0] = (bf16)f[0]; r[i + 1] = (bf16)f[1];
    r[i + 2] = (bf16)f[2]; r[i + 3] = (bf16)f[3];
  }
  return r;
}

static __device__ __forceinline__ v8f wmma_bf16(v16bf a, v16bf b, v8f c) {
  return __builtin_amdgcn_wmma_f32_16x16x32_bf16(false, a, false, b, (short)0, c,
                                                 false, false);
}

#if HAVE_TDM
// Build a 2D TDM descriptor (CDNA5 ISA 8.3/8.4):
//  grp0: count=1 | lds_addr | global_addr[56:0] | type=2
//  grp1: data_size (log2 bytes), tensor_dim0/1, tile_dim0/1, tensor_dim0_stride
static __device__ __forceinline__ void tdm_desc(u32x4* g0o, i32x8* g1o,
                                                unsigned long long ga,
                                                unsigned lds, int dsz_log,
                                                int tdim0, int tdim1,
                                                int stride0, int tile0,
                                                int tile1) {
  u32x4 g0;
  g0[0] = 1u;                                                 // count=1
  g0[1] = lds;                                                // lds_addr
  g0[2] = (unsigned)ga;                                       // gaddr[31:0]
  g0[3] = (unsigned)((ga >> 32) & 0x01ffffffu) | (2u << 30);  // gaddr hi|type=2
  i32x8 g1;
  g1[0] = (int)((unsigned)dsz_log << 16);                     // data_size
  g1[1] = (int)(((unsigned)tdim0 & 0xffffu) << 16);           // tdim0 lo
  g1[2] = (int)((((unsigned)tdim0 >> 16) & 0xffffu) |
                (((unsigned)tdim1 & 0xffffu) << 16));         // tdim0 hi|tdim1 lo
  g1[3] = (int)((((unsigned)tdim1 >> 16) & 0xffffu) |
                ((unsigned)tile0 << 16));                     // tdim1 hi|tile0
  g1[4] = tile1 & 0xffff;                                     // tile1, tile2=0
  g1[5] = stride0;                                            // stride0 lo
  g1[6] = 0;
  g1[7] = 0;
  *g0o = g0;
  *g1o = g1;
}

static __device__ __forceinline__ void tdm_load_tile2d(
    const void* gsrc, void* lds_dst, int dsz_log, int tdim0, int tdim1,
    int stride0, int tile0, int tile1) {
  u32x4 g0; i32x8 g1;
  tdm_desc(&g0, &g1, (unsigned long long)gsrc,
           (unsigned)(unsigned long long)lds_dst, dsz_log, tdim0, tdim1,
           stride0, tile0, tile1);
  i32x4 gz;
#pragma unroll
  for (int i = 0; i < 4; ++i) gz[i] = 0;
#if defined(__clang_major__) && (__clang_major__ >= 23)
  i32x8 gz8;
#pragma unroll
  for (int i = 0; i < 8; ++i) gz8[i] = 0;
  __builtin_amdgcn_tensor_load_to_lds(g0, g1, gz, gz, gz8, 0);
#else
  __builtin_amdgcn_tensor_load_to_lds(g0, g1, gz, gz, 0);
#endif
}

#if HAVE_TDM_STORE
static __device__ __forceinline__ void tdm_store_tile2d(
    void* gdst, const void* lds_src, int dsz_log, int tdim0, int tdim1,
    int stride0, int tile0, int tile1) {
  u32x4 g0; i32x8 g1;
  tdm_desc(&g0, &g1, (unsigned long long)gdst,
           (unsigned)(unsigned long long)lds_src, dsz_log, tdim0, tdim1,
           stride0, tile0, tile1);
  i32x4 gz;
#pragma unroll
  for (int i = 0; i < 4; ++i) gz[i] = 0;
#if defined(__clang_major__) && (__clang_major__ >= 23)
  i32x8 gz8;
#pragma unroll
  for (int i = 0; i < 8; ++i) gz8[i] = 0;
  __builtin_amdgcn_tensor_store_from_lds(g0, g1, gz, gz, gz8, 0);
#else
  __builtin_amdgcn_tensor_store_from_lds(g0, g1, gz, gz, 0);
#endif
}
#endif
#endif  // HAVE_TDM

// ---------------------------------------------------------------------------
// RMSNorm: f32 row (DIM) -> bf16 row
// ---------------------------------------------------------------------------
__global__ __launch_bounds__(256) void rmsnorm_kernel(const float* __restrict__ x,
                                                      const float* __restrict__ w,
                                                      bf16* __restrict__ out) {
  const int t = blockIdx.x;
  const float* row = x + (size_t)t * DIM;
  __shared__ float red[256];
  float ss = 0.f;
#pragma unroll
  for (int j = 0; j < DIM / 256; ++j) {
    float v = row[threadIdx.x + 256 * j];
    ss += v * v;
  }
  red[threadIdx.x] = ss;
  __syncthreads();
  for (int off = 128; off; off >>= 1) {
    if (threadIdx.x < off) red[threadIdx.x] += red[threadIdx.x + off];
    __syncthreads();
  }
  const float inv = rsqrtf(red[0] / (float)DIM + 1e-6f);
#pragma unroll
  for (int j = 0; j < DIM / 256; ++j) {
    int idx = threadIdx.x + 256 * j;
    out[(size_t)t * DIM + idx] = (bf16)(row[idx] * inv * w[idx]);
  }
}

// ---------------------------------------------------------------------------
// WMMA GEMM: C[M,N] = A[M,K](bf16) @ W[N,K](f32)^T  (+ residual R)
// 8 waves share one 16-row A tile staged to LDS by the TDM (double buffered).
// ---------------------------------------------------------------------------
__global__ __launch_bounds__(256) void gemm_kernel(const bf16* __restrict__ A,
                                                   const float* __restrict__ W,
                                                   const float* __restrict__ R,
                                                   float* __restrict__ C,
                                                   int M, int N, int K) {
  const int wave = threadIdx.x >> 5;
  const int lane = threadIdx.x & 31;
  const int mrow = lane & 15, khalf = lane >> 4;
  const int nt = blockIdx.x * 8 + wave;
  const int mt = blockIdx.y;
  const bool active = ((nt + 1) * 16 <= N);
  const int ntc = active ? nt : 0;

  __shared__ __align__(32) bf16 atile[2][16 * AK_CHUNK];  // 2 x 8KB

  const bf16* asrc = A + (size_t)(mt * 16) * K;
  const float* wrow = W + (size_t)(ntc * 16 + mrow) * K + khalf * 16;
  const int nchunk = K / AK_CHUNK;

#if HAVE_TDM
  if (wave == 0)
    tdm_load_tile2d(asrc, &atile[0][0], 1, K, M, K, AK_CHUNK, 16);
#else
  for (int idx = threadIdx.x; idx < 16 * AK_CHUNK / 8; idx += 256) {
    int r = idx / (AK_CHUNK / 8), c8 = idx % (AK_CHUNK / 8);
    *(v8bf*)&atile[0][r * AK_CHUNK + c8 * 8] =
        *(const v8bf*)(asrc + (size_t)r * K + c8 * 8);
  }
#endif

  v8f c = zero8();
  for (int i = 0; i < nchunk; ++i) {
    if (i + 1 < nchunk) {
#if HAVE_TDM
      if (wave == 0)
        tdm_load_tile2d(asrc + (size_t)(i + 1) * AK_CHUNK, &atile[(i + 1) & 1][0],
                        1, K, M, K, AK_CHUNK, 16);
#else
      const bf16* s2 = asrc + (size_t)(i + 1) * AK_CHUNK;
      for (int idx = threadIdx.x; idx < 16 * AK_CHUNK / 8; idx += 256) {
        int r = idx / (AK_CHUNK / 8), c8 = idx % (AK_CHUNK / 8);
        *(v8bf*)&atile[(i + 1) & 1][r * AK_CHUNK + c8 * 8] =
            *(const v8bf*)(s2 + (size_t)r * K + c8 * 8);
      }
#endif
    }
#if HAVE_TDM
    if (wave == 0) {
      if (i + 1 < nchunk) __builtin_amdgcn_s_wait_tensorcnt(1);
      else                __builtin_amdgcn_s_wait_tensorcnt(0);
    }
#endif
    __syncthreads();
    if (active) {  // wave-uniform: EXEC stays all-ones for WMMA
      const bf16* abase = &atile[i & 1][0];
#pragma unroll
      for (int k0 = 0; k0 < AK_CHUNK; k0 += 32) {
        __builtin_prefetch(wrow + i * AK_CHUNK + k0 + 256, 0, 1);
        v16bf a = load_afrag(abase + mrow * AK_CHUNK + k0, khalf);
        v16bf b = load_wfrag(wrow + i * AK_CHUNK + k0);
        c = wmma_bf16(a, b, c);
      }
    }
    __syncthreads();
  }
  if (active) {
#pragma unroll
    for (int r = 0; r < 8; ++r) {
      size_t idx = (size_t)(mt * 16 + r + 8 * khalf) * N + nt * 16 + mrow;
      float v = c[r];
      if (R) v += R[idx];
      C[idx] = v;
    }
  }
}

// ---------------------------------------------------------------------------
// Per-head RMSNorm + RoPE for Q and K; writes bf16
// ---------------------------------------------------------------------------
__global__ __launch_bounds__(128) void qknorm_rope_kernel(
    const float* __restrict__ qf, const float* __restrict__ kf,
    const float* __restrict__ cosp, const float* __restrict__ sinp,
    const float* __restrict__ qnw, const float* __restrict__ knw,
    bf16* __restrict__ q_bf, bf16* __restrict__ k_bf) {
  const int t = blockIdx.x;
  const int hh = blockIdx.y;
  const int d = threadIdx.x;
  const bool isq = (hh < NH);
  const float* src = isq ? (qf + ((size_t)t * NH + hh) * HD)
                         : (kf + ((size_t)t * NKV + (hh - NH)) * HD);
  const float* nw = isq ? qnw : knw;
  __shared__ float red[HD];
  __shared__ float buf[HD];
  float xv = src[d];
  red[d] = xv * xv;
  __syncthreads();
  for (int off = 64; off; off >>= 1) {
    if (d < off) red[d] += red[d + off];
    __syncthreads();
  }
  const float inv = rsqrtf(red[0] / (float)HD + 1e-6f);
  const float xn = xv * inv * nw[d];
  buf[d] = xn;
  __syncthreads();
  const float rot = (d < HD / 2) ? -buf[d + HD / 2] : buf[d - HD / 2];
  const float o = xn * cosp[(size_t)t * HD + d] + rot * sinp[(size_t)t * HD + d];
  if (isq) q_bf[((size_t)t * NH + hh) * HD + d] = (bf16)o;
  else     k_bf[((size_t)t * NKV + (hh - NH)) * HD + d] = (bf16)o;
}

// V: f32 [S][NKV][HD] -> bf16 transposed [NKV][HD][S]
__global__ __launch_bounds__(256) void vcvt_kernel(const float* __restrict__ v,
                                                   bf16* __restrict__ vT) {
  int idx = blockIdx.x * 256 + threadIdx.x;
  if (idx >= NKV * HD * S_LEN) return;
  int s = idx & (S_LEN - 1);
  int d = (idx >> 9) & (HD - 1);
  int kvh = idx >> 16;
  vT[idx] = (bf16)(v[(size_t)s * (NKV * HD) + kvh * HD + d]);
}

// ---------------------------------------------------------------------------
// Flash attention: one wave per 16-query tile, 32 keys per step (full K=32 PV)
// ---------------------------------------------------------------------------
__global__ __launch_bounds__(128) void attn_kernel(const bf16* __restrict__ qbp,
                                                   const bf16* __restrict__ kbp,
                                                   const bf16* __restrict__ vtp,
                                                   bf16* __restrict__ ob) {
  const int h = blockIdx.x;
  const int wave = threadIdx.x >> 5;
  const int lane = threadIdx.x & 31;
  const int qt = blockIdx.y * 4 + wave;
  const int q0 = qt * 16;
  const int kvh = h >> 3;
  const int mrow = lane & 15, khalf = lane >> 4;

  __shared__ __align__(32) bf16 ldsP[4][16 * 32];
  bf16* Pw = &ldsP[wave][0];

  v16bf qfrag[4];
  const bf16* qrow = qbp + (size_t)(q0 + mrow) * (NH * HD) + h * HD;
#pragma unroll
  for (int kc = 0; kc < 4; ++kc) qfrag[kc] = load_afrag(qrow + kc * 32, khalf);

  v8f o[8];
#pragma unroll
  for (int d = 0; d < 8; ++d) o[d] = zero8();
  float m_i[8], l_i[8];
#pragma unroll
  for (int r = 0; r < 8; ++r) { m_i[r] = -1e30f; l_i[r] = 0.f; }

  const int nkb = ((q0 + 15) >> 5) + 1;
  for (int kb = 0; kb < nkb; ++kb) {
    const int k0 = kb * 32;
    v8f sa = zero8(), sb = zero8();
    const bf16* krowa = kbp + (size_t)(k0 + mrow) * (NKV * HD) + kvh * HD;
    const bf16* krowb = krowa + (size_t)16 * (NKV * HD);
#pragma unroll
    for (int kc = 0; kc < 4; ++kc) {
      v16bf ka = load_bfrag(krowa + kc * 32 + khalf * 16);
      v16bf kb2 = load_bfrag(krowb + kc * 32 + khalf * 16);
      sa = wmma_bf16(qfrag[kc], ka, sa);
      sb = wmma_bf16(qfrag[kc], kb2, sb);
    }
    const float scl = 0.08838834764831845f;  // 1/sqrt(128)
#pragma unroll
    for (int r = 0; r < 8; ++r) {
      const int qq = q0 + r + 8 * khalf;
      float va = (k0 + mrow      <= qq) ? sa[r] * scl : -1e30f;
      float vb = (k0 + 16 + mrow <= qq) ? sb[r] * scl : -1e30f;
      float mx = fmaxf(va, vb);
#pragma unroll
      for (int off = 8; off; off >>= 1) mx = fmaxf(mx, __shfl_xor(mx, off, 16));
      const float mnew = fmaxf(m_i[r], mx);
      const float ea = __expf(va - mnew);
      const float eb = __expf(vb - mnew);
      float rs = ea + eb;
#pragma unroll
      for (int off = 8; off; off >>= 1) rs += __shfl_xor(rs, off, 16);
      const float cor = __expf(m_i[r] - mnew);
      l_i[r] = l_i[r] * cor + rs;
      m_i[r] = mnew;
#pragma unroll
      for (int d = 0; d < 8; ++d) o[d][r] = o[d][r] * cor;
      const int m = r + 8 * khalf;
      Pw[m * 32 + mrow] = (bf16)ea;
      Pw[m * 32 + 16 + mrow] = (bf16)eb;
    }
    // cross-lane LDS exchange: wait on DS stores before A-fragment reads
    asm volatile("s_wait_dscnt 0" ::: "memory");
    v16bf pf = load_afrag(Pw + mrow * 32, khalf);
#pragma unroll
    for (int d = 0; d < 8; ++d) {
      const bf16* vrow =
          vtp + ((size_t)kvh * HD + d * 16 + mrow) * S_LEN + k0 + khalf * 16;
      o[d] = wmma_bf16(pf, load_bfrag(vrow), o[d]);
    }
  }
#pragma unroll
  for (int r = 0; r < 8; ++r) {
    const float inv = 1.f / l_i[r];
    const int m = q0 + r + 8 * khalf;
#pragma unroll
    for (int d = 0; d < 8; ++d)
      ob[(size_t)m * (NH * HD) + h * HD + d * 16 + mrow] = (bf16)(o[d][r] * inv);
  }
}

// ---------------------------------------------------------------------------
// Router: softmax over 64 experts, top-8, scatter (token, weight) per expert
// ---------------------------------------------------------------------------
__global__ __launch_bounds__(64) void router_kernel(const float* __restrict__ logits,
                                                    int* __restrict__ cnt,
                                                    int* __restrict__ tok,
                                                    float* __restrict__ wgt) {
  const int t = blockIdx.x;
  const int i = threadIdx.x;
  __shared__ float red[NE];
  __shared__ float sl[NE];
  float v = logits[(size_t)t * NE + i];
  red[i] = v;
  __syncthreads();
  for (int off = 32; off; off >>= 1) {
    if (i < off) red[i] = fmaxf(red[i], red[i + off]);
    __syncthreads();
  }
  const float mx = red[0];
  __syncthreads();
  const float ex = __expf(v - mx);
  red[i] = ex;
  __syncthreads();
  for (int off = 32; off; off >>= 1) {
    if (i < off) red[i] += red[i + off];
    __syncthreads();
  }
  sl[i] = ex / red[0];
  __syncthreads();
  if (i == 0) {
    int ti[NTOP];
    float tw[NTOP];
    float tot = 0.f;
    unsigned long long used = 0ull;
    for (int j = 0; j < NTOP; ++j) {
      float best = -1.f;
      int bi = 0;
      for (int e = 0; e < NE; ++e)
        if (!((used >> e) & 1ull) && sl[e] > best) { best = sl[e]; bi = e; }
      used |= (1ull << bi);
      ti[j] = bi;
      tw[j] = best;
      tot += best;
    }
    for (int j = 0; j < NTOP; ++j) {
      const float w = tw[j] / (tot + 1e-20f);
      const int e = ti[j];
      const int idx = atomicAdd(&cnt[e], 1);
      tok[e * S_LEN + idx] = t;
      wgt[e * S_LEN + idx] = w;
    }
  }
}

// ---------------------------------------------------------------------------
// Residual copy h -> out. TDM path: DMA 16x512 f32 tiles through LDS without
// touching VGPRs (tensor_load_to_lds + tensor_store_from_lds, TENSORcnt).
// grid = S/16 blocks of a single wave.
// ---------------------------------------------------------------------------
__global__ __launch_bounds__(32) void copy_tdm_kernel(const float* __restrict__ s,
                                                      float* __restrict__ d) {
#if HAVE_TDM && HAVE_TDM_STORE
  __shared__ __align__(16) float buf[16 * 512];  // 32KB
  const int r0 = blockIdx.x * 16;
  for (int c0 = 0; c0 < DIM; c0 += 512) {
    tdm_load_tile2d(s + (size_t)r0 * DIM + c0, buf, 2, DIM, S_LEN, DIM, 512, 16);
    __builtin_amdgcn_s_wait_tensorcnt(0);
    tdm_store_tile2d(d + (size_t)r0 * DIM + c0, buf, 2, DIM, S_LEN, DIM, 512, 16);
    __builtin_amdgcn_s_wait_tensorcnt(0);
  }
#else
  const size_t base = (size_t)blockIdx.x * 16 * DIM;
  for (int idx = threadIdx.x; idx < 16 * DIM; idx += 32)
    d[base + idx] = s[base + idx];
#endif
}

// ---------------------------------------------------------------------------
// Fused MoE expert FFN: per (expert, 32-token tile):
//   mid = silu(A @ gate^T) * (A @ up^T)  -> LDS (bf16)
//   y   = mid @ down^T ; out[token] += w * y   (global f32 atomics)
// ---------------------------------------------------------------------------
__global__ __launch_bounds__(256) void moe_kernel(
    const bf16* __restrict__ hn_bf, const float* __restrict__ gate_w,
    const float* __restrict__ up_w, const float* __restrict__ down_w,
    const int* __restrict__ cnt, const int* __restrict__ tok,
    const float* __restrict__ wgt, float* __restrict__ out) {
  const int e = blockIdx.x;
  const int tile = blockIdx.y;
  const int c = cnt[e];
  const int base = tile * 32;
  if (base >= c) return;
  const int rows = min(32, c - base);

  __shared__ __align__(32) bf16 mid[32 * FF];  // 48KB (of 320KB/WGP)
  __shared__ int stok[32];
  __shared__ float swgt[32];
  if (threadIdx.x < 32) {
    const int li = base + threadIdx.x;
    stok[threadIdx.x] = (threadIdx.x < rows) ? tok[e * S_LEN + li] : 0;
    swgt[threadIdx.x] = (threadIdx.x < rows) ? wgt[e * S_LEN + li] : 0.f;
  }
  __syncthreads();

  const int wave = threadIdx.x >> 5;
  const int lane = threadIdx.x & 31;
  const int mrow = lane & 15, khalf = lane >> 4;

  // ---- phase 1: gate & up projections (M=32, N=FF, K=DIM) ----
  for (int p = wave; p < 2 * (FF / 16); p += 8) {
    const int mt = p & 1, nt = p >> 1;
    const bf16* arow = hn_bf + (size_t)stok[mt * 16 + mrow] * DIM;
    const float* grow = gate_w + ((size_t)e * FF + nt * 16 + mrow) * DIM + khalf * 16;
    const float* urow = up_w + ((size_t)e * FF + nt * 16 + mrow) * DIM + khalf * 16;
    v8f cg = zero8(), cu = zero8();
    for (int k0 = 0; k0 < DIM; k0 += 32) {
      __builtin_prefetch(grow + k0 + 256, 0, 1);
      v16bf a = load_afrag(arow + k0, khalf);
      cg = wmma_bf16(a, load_wfrag(grow + k0), cg);
      cu = wmma_bf16(a, load_wfrag(urow + k0), cu);
    }
#pragma unroll
    for (int r = 0; r < 8; ++r) {
      const float g = cg[r], u = cu[r];
      const float midv = (g / (1.f + __expf(-g))) * u;  // silu(g)*u
      const int m = mt * 16 + r + 8 * khalf;
      const int n = nt * 16 + mrow;
      mid[m * FF + n] = (bf16)midv;
    }
  }
  __syncthreads();

  // ---- phase 2: down projection (M=32, N=DIM, K=FF) + weighted scatter ----
  for (int p = wave; p < 2 * (DIM / 16); p += 8) {
    const int mt = p & 1, nt = p >> 1;
    const bf16* arow = &mid[(mt * 16 + mrow) * FF];
    const float* drow = down_w + ((size_t)e * DIM + nt * 16 + mrow) * FF + khalf * 16;
    v8f cy = zero8();
    for (int k0 = 0; k0 < FF; k0 += 32) {
      v16bf a = load_afrag(arow + k0, khalf);
      cy = wmma_bf16(a, load_wfrag(drow + k0), cy);
    }
#pragma unroll
    for (int r = 0; r < 8; ++r) {
      const int m = mt * 16 + r + 8 * khalf;
      if (m < rows) {
        const int col = nt * 16 + mrow;
        atomicAdd(&out[(size_t)stok[m] * DIM + col], cy[r] * swgt[m]);
      }
    }
  }
}

// ---------------------------------------------------------------------------
extern "C" void kernel_launch(void* const* d_in, const int* in_sizes, int n_in,
                              void* d_out, int out_size, void* d_ws, size_t ws_size,
                              hipStream_t stream) {
  (void)in_sizes; (void)n_in; (void)out_size; (void)ws_size;
  const float* x    = (const float*)d_in[0];
  const float* cosp = (const float*)d_in[1];
  const float* sinp = (const float*)d_in[2];
  const float* n1w  = (const float*)d_in[3];
  const float* wq   = (const float*)d_in[4];
  const float* wk   = (const float*)d_in[5];
  const float* wv   = (const float*)d_in[6];
  const float* wo   = (const float*)d_in[7];
  const float* qnw  = (const float*)d_in[8];
  const float* knw  = (const float*)d_in[9];
  const float* n2w  = (const float*)d_in[10];
  const float* rw   = (const float*)d_in[11];
  const float* gw   = (const float*)d_in[12];
  const float* uw   = (const float*)d_in[13];
  const float* dw   = (const float*)d_in[14];
  float* out = (float*)d_out;

  char* p = (char*)d_ws;
  auto alloc = [&](size_t bytes) -> void* {
    void* r = (void*)p;
    p += (bytes + 255) & ~(size_t)255;
    return r;
  };
  bf16*  xn_bf   = (bf16*)alloc((size_t)S_LEN * DIM * 2);
  float* qf      = (float*)alloc((size_t)S_LEN * NH * HD * 4);
  float* kf      = (float*)alloc((size_t)S_LEN * NKV * HD * 4);
  float* vf      = (float*)alloc((size_t)S_LEN * NKV * HD * 4);
  bf16*  q_bf    = (bf16*)alloc((size_t)S_LEN * NH * HD * 2);
  bf16*  k_bf    = (bf16*)alloc((size_t)S_LEN * NKV * HD * 2);
  bf16*  vT      = (bf16*)alloc((size_t)S_LEN * NKV * HD * 2);
  bf16*  attn_bf = (bf16*)alloc((size_t)S_LEN * NH * HD * 2);
  float* h       = (float*)alloc((size_t)S_LEN * DIM * 4);
  bf16*  hn_bf   = (bf16*)alloc((size_t)S_LEN * DIM * 2);
  float* logits  = (float*)alloc((size_t)S_LEN * NE * 4);
  int*   cnt     = (int*)alloc((size_t)NE * 4);
  int*   tok     = (int*)alloc((size_t)NE * S_LEN * 4);
  float* wgt     = (float*)alloc((size_t)NE * S_LEN * 4);

  hipMemsetAsync(cnt, 0, NE * sizeof(int), stream);

  rmsnorm_kernel<<<S_LEN, 256, 0, stream>>>(x, n1w, xn_bf);

  gemm_kernel<<<dim3((NH * HD) / 128, S_LEN / 16), 256, 0, stream>>>(
      xn_bf, wq, nullptr, qf, S_LEN, NH * HD, DIM);
  gemm_kernel<<<dim3((NKV * HD) / 128, S_LEN / 16), 256, 0, stream>>>(
      xn_bf, wk, nullptr, kf, S_LEN, NKV * HD, DIM);
  gemm_kernel<<<dim3((NKV * HD) / 128, S_LEN / 16), 256, 0, stream>>>(
      xn_bf, wv, nullptr, vf, S_LEN, NKV * HD, DIM);

  qknorm_rope_kernel<<<dim3(S_LEN, NH + NKV), 128, 0, stream>>>(
      qf, kf, cosp, sinp, qnw, knw, q_bf, k_bf);
  vcvt_kernel<<<(NKV * HD * S_LEN + 255) / 256, 256, 0, stream>>>(vf, vT);

  attn_kernel<<<dim3(NH, S_LEN / 16 / 4), 128, 0, stream>>>(q_bf, k_bf, vT, attn_bf);

  gemm_kernel<<<dim3(DIM / 128, S_LEN / 16), 256, 0, stream>>>(
      attn_bf, wo, x, h, S_LEN, DIM, NH * HD);  // + residual x

  rmsnorm_kernel<<<S_LEN, 256, 0, stream>>>(h, n2w, hn_bf);

  gemm_kernel<<<dim3(1, S_LEN / 16), 256, 0, stream>>>(
      hn_bf, rw, nullptr, logits, S_LEN, NE, DIM);
  router_kernel<<<S_LEN, NE, 0, stream>>>(logits, cnt, tok, wgt);

  copy_tdm_kernel<<<S_LEN / 16, 32, 0, stream>>>(h, out);
  moe_kernel<<<dim3(NE, S_LEN / 32), 256, 0, stream>>>(hn_bf, gw, uw, dw, cnt, tok,
                                                       wgt, out);
}
